// SparseMoE_26414048870706
// MI455X (gfx1250) — matrix operand order repaired and checked by reading, not compiled, not run on hardware
//
#include <hip/hip_runtime.h>
#include <hip/hip_bf16.h>
#include <math.h>

// ---------------- problem constants ----------------
#define NTOK  32768      // B*T
#define NE    8          // experts
#define DDIM  512
#define HDIM  2048       // 4*D
#define CAP   8192       // NTOK*TOPK/NE
#define HC    256        // hidden chunk staged in LDS
#define MT    64         // row tile per workgroup
#define LDSS  264        // padded LDS row stride (bf16 elems): 528B row, 16B aligned, bank-skewed

typedef __bf16 bf16;
typedef __attribute__((ext_vector_type(16))) __bf16 v16bf;
typedef __attribute__((ext_vector_type(8)))  float  v8f;

union FragAB { v16bf v; uint4 q[2]; };
union Pack8  { bf16 h[8]; uint4 q; };

// ---------------- 1) router: noisy top-2 ----------------
__global__ __launch_bounds__(256)
void moe_router(const float* __restrict__ x, const float* __restrict__ noise,
                const float* __restrict__ Ww, const float* __restrict__ bw,
                const float* __restrict__ Wn, const float* __restrict__ bn,
                int* __restrict__ ix0, int* __restrict__ ix1,
                float* __restrict__ p0, float* __restrict__ p1,
                int* __restrict__ ts0, int* __restrict__ ts1)
{
    const int w = threadIdx.x >> 5, lane = threadIdx.x & 31;
    const int n = blockIdx.x * 8 + w;            // one wave32 per token
    const float* xr = x + (size_t)n * DDIM;

    float aw[NE], an[NE];
#pragma unroll
    for (int e = 0; e < NE; ++e) { aw[e] = 0.f; an[e] = 0.f; }
    for (int kk = 0; kk < DDIM / 32; ++kk) {
        const int k = kk * 32 + lane;
        const float xv = xr[k];
#pragma unroll
        for (int e = 0; e < NE; ++e) {
            aw[e] = fmaf(xv, Ww[e * DDIM + k], aw[e]);
            an[e] = fmaf(xv, Wn[e * DDIM + k], an[e]);
        }
    }
#pragma unroll
    for (int e = 0; e < NE; ++e)
        for (int m = 16; m >= 1; m >>= 1) {
            aw[e] += __shfl_xor(aw[e], m, 32);
            an[e] += __shfl_xor(an[e], m, 32);
        }
    if (lane == 0) {
        float nv[NE];
#pragma unroll
        for (int e = 0; e < NE; ++e) {
            const float lg  = aw[e] + bw[e];
            const float pre = an[e] + bn[e];
            const float sp  = fmaxf(pre, 0.f) + log1pf(expf(-fabsf(pre)));  // softplus
            nv[e] = lg + noise[(size_t)n * NE + e] * sp;
        }
        float b1v = -INFINITY; int bi1 = 0;
#pragma unroll
        for (int e = 0; e < NE; ++e) if (nv[e] > b1v) { b1v = nv[e]; bi1 = e; }
        float b2v = -INFINITY; int bi2 = 0;
#pragma unroll
        for (int e = 0; e < NE; ++e) if (e != bi1 && nv[e] > b2v) { b2v = nv[e]; bi2 = e; }
        const float e1 = expf(b2v - b1v);        // <= 1, stable
        const float q0 = 1.f / (1.f + e1);
        ix0[n] = bi1; ix1[n] = bi2;
        p0[n] = q0;  p1[n] = 1.f - q0;
        ts0[n] = -1; ts1[n] = -1;
    }
}

// ---------------- 2) stable per-expert compaction (order-preserving) ----------------
__global__ __launch_bounds__(1024)
void moe_scan(const int* __restrict__ ix0, const int* __restrict__ ix1,
              const float* __restrict__ p0, const float* __restrict__ p1,
              int* __restrict__ etok, float* __restrict__ egate,
              int* __restrict__ ts0, int* __restrict__ ts1, int* __restrict__ cnt)
{
    const int e = blockIdx.x, tid = threadIdx.x;
    __shared__ int sd[1024];
    int base = 0;
    for (int it = 0; it < NTOK / 1024; ++it) {
        const int tok = it * 1024 + tid;
        const int i0 = ix0[tok], i1 = ix1[tok];
        const int m = (i0 == e) || (i1 == e);
        sd[tid] = m;
        __syncthreads();
        for (int off = 1; off < 1024; off <<= 1) {     // Hillis-Steele inclusive scan
            const int v = sd[tid];
            const int add = (tid >= off) ? sd[tid - off] : 0;
            __syncthreads();
            sd[tid] = v + add;
            __syncthreads();
        }
        const int incl = sd[tid];
        const int tot  = sd[1023];
        if (m) {
            const int pos = base + incl - 1;
            if (pos < CAP) {
                const int gs = e * CAP + pos;
                etok[gs]  = tok;
                egate[gs] = (i0 == e) ? p0[tok] : p1[tok];
                if (i0 == e) ts0[tok] = gs; else ts1[tok] = gs;
            }
        }
        base += tot;
        __syncthreads();
    }
    if (tid == 0) cnt[e] = base < CAP ? base : CAP;
}

// ---------------- 3) weights f32 -> bf16, 8 elems/thread ----------------
__global__ __launch_bounds__(256)
void moe_cvtw(const float* __restrict__ W1, const float* __restrict__ W2,
              bf16* __restrict__ W1b, bf16* __restrict__ W2b)
{
    const size_t i = ((size_t)blockIdx.x * 256 + threadIdx.x) * 8;
    const float4 a0 = *(const float4*)(W1 + i);
    const float4 a1 = *(const float4*)(W1 + i + 4);
    const float4 c0 = *(const float4*)(W2 + i);
    const float4 c1 = *(const float4*)(W2 + i + 4);
    Pack8 oa, oc;
    oa.h[0] = (bf16)a0.x; oa.h[1] = (bf16)a0.y; oa.h[2] = (bf16)a0.z; oa.h[3] = (bf16)a0.w;
    oa.h[4] = (bf16)a1.x; oa.h[5] = (bf16)a1.y; oa.h[6] = (bf16)a1.z; oa.h[7] = (bf16)a1.w;
    oc.h[0] = (bf16)c0.x; oc.h[1] = (bf16)c0.y; oc.h[2] = (bf16)c0.z; oc.h[3] = (bf16)c0.w;
    oc.h[4] = (bf16)c1.x; oc.h[5] = (bf16)c1.y; oc.h[6] = (bf16)c1.z; oc.h[7] = (bf16)c1.w;
    *(uint4*)(W1b + i) = oa.q;
    *(uint4*)(W2b + i) = oc.q;
}

// ---------------- 4) gather token rows -> bf16 slot buffer (2 rows/block) ----------------
__global__ __launch_bounds__(128)
void moe_gather_x(const float* __restrict__ x, const int* __restrict__ etok,
                  const int* __restrict__ cnt, bf16* __restrict__ Xg)
{
    const int slot = blockIdx.x * 2 + (threadIdx.x >> 6);
    const int t = threadIdx.x & 63;
    const int e = slot >> 13, s = slot & (CAP - 1);
    if (s < cnt[e]) {
        const int tok = etok[slot];
        const float* src = x + (size_t)tok * DDIM + t * 8;
        const float4 a0 = *(const float4*)(src);
        const float4 a1 = *(const float4*)(src + 4);
        Pack8 o;
        o.h[0] = (bf16)a0.x; o.h[1] = (bf16)a0.y; o.h[2] = (bf16)a0.z; o.h[3] = (bf16)a0.w;
        o.h[4] = (bf16)a1.x; o.h[5] = (bf16)a1.y; o.h[6] = (bf16)a1.z; o.h[7] = (bf16)a1.w;
        *(uint4*)(Xg + (size_t)slot * DDIM + t * 8) = o.q;
    }
}

// ---------------- 5) fused FFN: relu(Xg@W1^T+b1)@W2^T+b2, gated ----------------
// 512 threads = 16 waves; tile = 64 rows x 512 cols, H chunked by 256 through LDS.
// wave: wr = w&1 (32-row half), wc = w>>1 (8 col groups).
// Each B fragment feeds 2 WMMAs (2 A-row fragments) to halve B-load traffic.
// The A-row OFFSET (not pointer) is laundered through inline asm each chunk: keeps
// global address-space provenance (global_load_b128, no flat/DScnt coupling) while
// still preventing LLVM from hoisting chunk-invariant A loads (scratch-spill bug).
__global__ __launch_bounds__(512)
void moe_ffn(const bf16* __restrict__ Xg, const bf16* __restrict__ W1b,
             const bf16* __restrict__ W2b, const float* __restrict__ b1,
             const float* __restrict__ b2, const float* __restrict__ egate,
             const int* __restrict__ cnt, float* __restrict__ outSlots)
{
    const int e = blockIdx.x >> 7;                 // 128 tiles/expert
    const int tileBase = (blockIdx.x & 127) * MT;
    if (tileBase >= cnt[e]) return;                // uniform early exit

    const int tid = threadIdx.x;
    const int w = tid >> 5, lane = tid & 31;
    const int l16 = lane & 15, half = lane >> 4;
    const int wr = w & 1;                          // 2 row groups of 32
    const int wc = w >> 1;                         // 8 col groups

    __shared__ bf16 hbuf[MT * LDSS];               // 64 x 256 hidden chunk (padded)

    const bf16* W1e = W1b + (size_t)e * HDIM * DDIM;
    const bf16* W2e = W2b + (size_t)e * DDIM * HDIM;
    const unsigned long long aoff =
        (unsigned long long)((size_t)(e * CAP + tileBase + wr * 32 + l16) * DDIM);

    v8f acc[2][4];                                 // 32 rows x 64 out cols
#pragma unroll
    for (int r = 0; r < 2; ++r)
#pragma unroll
        for (int c = 0; c < 4; ++c)
#pragma unroll
            for (int v = 0; v < 8; ++v) acc[r][c][v] = 0.f;

    for (int hc = 0; hc < HDIM / HC; ++hc) {
        const int hBase = hc * HC;
        // Opaque per-iteration copy of the A-row offset: blocks cross-iteration
        // CSE/hoist of A loads while keeping global address space on Xg.
        unsigned long long ao = aoff;
        asm volatile("" : "+v"(ao));
        const bf16* Ar0 = Xg + ao;
        const bf16* Ar1 = Ar0 + (size_t)16 * DDIM;

        // ---- GEMM1: h_chunk[64 x 256] = relu(x_tile @ W1_chunk^T + b1) ----
        v8f hacc[2][2];                            // 32 rows x 32 h cols
#pragma unroll
        for (int r = 0; r < 2; ++r)
#pragma unroll
            for (int c = 0; c < 2; ++c)
#pragma unroll
                for (int v = 0; v < 8; ++v) hacc[r][c][v] = 0.f;

#pragma unroll
        for (int ks = 0; ks < DDIM / 32; ++ks) {
            const int kb = ks * 32;
            FragAB a0, a1;                         // A layout: two 16B runs per lane
            a0.q[0] = *(const uint4*)(Ar0 + kb + half * 8);
            a0.q[1] = *(const uint4*)(Ar0 + kb + 16 + half * 8);
            a1.q[0] = *(const uint4*)(Ar1 + kb + half * 8);
            a1.q[1] = *(const uint4*)(Ar1 + kb + 16 + half * 8);
#pragma unroll
            for (int c = 0; c < 2; ++c) {
                const int nidx = hBase + wc * 32 + c * 16 + l16;
                const bf16* Bp = W1e + (size_t)nidx * DDIM + kb + half * 8;
                FragAB b;                          // B layout: two 16B runs per lane
                b.q[0] = *(const uint4*)(Bp);
                b.q[1] = *(const uint4*)(Bp + 16);
                hacc[0][c] = __builtin_amdgcn_wmma_f32_16x16x32_bf16(
                    false, a0.v, false, b.v, (short)0, hacc[0][c], false, false);
                hacc[1][c] = __builtin_amdgcn_wmma_f32_16x16x32_bf16(
                    false, a1.v, false, b.v, (short)0, hacc[1][c], false, false);
            }
        }
#pragma unroll
        for (int c = 0; c < 2; ++c) {
            const int col = wc * 32 + c * 16 + l16;
            const float bias = b1[e * HDIM + hBase + col];
#pragma unroll
            for (int r = 0; r < 2; ++r)
#pragma unroll
                for (int v = 0; v < 8; ++v) {      // C/D layout: M = v + half*8
                    const float hv = fmaxf(hacc[r][c][v] + bias, 0.f);
                    hbuf[(wr * 32 + r * 16 + v + half * 8) * LDSS + col] = (bf16)hv;
                }
        }
        __syncthreads();
        // ---- GEMM2: out[64 x 512] += h_chunk @ W2_chunk^T ----
#pragma unroll
        for (int ks = 0; ks < HC / 32; ++ks) {
            const int kb = ks * 32;
            FragAB a0, a1;
            const bf16* hp0 = &hbuf[(wr * 32 + l16) * LDSS + kb + half * 8];
            const bf16* hp1 = hp0 + 16 * LDSS;
            a0.q[0] = *(const uint4*)(hp0);
            a0.q[1] = *(const uint4*)(hp0 + 16);
            a1.q[0] = *(const uint4*)(hp1);
            a1.q[1] = *(const uint4*)(hp1 + 16);
#pragma unroll
            for (int c = 0; c < 4; ++c) {
                const int nidx = wc * 64 + c * 16 + l16;
                const bf16* Bp = W2e + (size_t)nidx * HDIM + hBase + kb + half * 8;
                FragAB b;
                b.q[0] = *(const uint4*)(Bp);
                b.q[1] = *(const uint4*)(Bp + 16);
                acc[0][c] = __builtin_amdgcn_wmma_f32_16x16x32_bf16(
                    false, a0.v, false, b.v, (short)0, acc[0][c], false, false);
                acc[1][c] = __builtin_amdgcn_wmma_f32_16x16x32_bf16(
                    false, a1.v, false, b.v, (short)0, acc[1][c], false, false);
            }
        }
        __syncthreads();
    }
    // epilogue: + b2, scale by gate, write per-slot rows
#pragma unroll
    for (int r = 0; r < 2; ++r)
#pragma unroll
        for (int v = 0; v < 8; ++v) {
            const int row = tileBase + wr * 32 + r * 16 + v + half * 8;
            const float g = egate[e * CAP + row];
            float* orow = outSlots + (size_t)(e * CAP + row) * DDIM;
#pragma unroll
            for (int c = 0; c < 4; ++c) {
                const int nidx = wc * 64 + c * 16 + l16;
                orow[nidx] = (acc[r][c][v] + b2[e * DDIM + nidx]) * g;
            }
        }
}

// ---------------- 6) per-token gather of <=2 slot rows (float4) ----------------
__global__ __launch_bounds__(128)
void moe_gather_out(const float* __restrict__ outSlots, const int* __restrict__ ts0,
                    const int* __restrict__ ts1, float* __restrict__ out)
{
    const int n = blockIdx.x;
    const int s0 = ts0[n], s1 = ts1[n];
    const int idx = threadIdx.x * 4;
    float4 v = make_float4(0.f, 0.f, 0.f, 0.f);
    if (s0 >= 0) {
        const float4 a = *(const float4*)(outSlots + (size_t)s0 * DDIM + idx);
        v.x += a.x; v.y += a.y; v.z += a.z; v.w += a.w;
    }
    if (s1 >= 0) {
        const float4 a = *(const float4*)(outSlots + (size_t)s1 * DDIM + idx);
        v.x += a.x; v.y += a.y; v.z += a.z; v.w += a.w;
    }
    *(float4*)(out + (size_t)n * DDIM + idx) = v;
}

// ---------------- host launcher ----------------
extern "C" void kernel_launch(void* const* d_in, const int* in_sizes, int n_in,
                              void* d_out, int out_size, void* d_ws, size_t ws_size,
                              hipStream_t stream)
{
    (void)in_sizes; (void)n_in; (void)out_size; (void)ws_size;
    const float* x     = (const float*)d_in[0];
    const float* noise = (const float*)d_in[1];
    const float* Ww    = (const float*)d_in[2];
    const float* bw    = (const float*)d_in[3];
    const float* Wn    = (const float*)d_in[4];
    const float* bn    = (const float*)d_in[5];
    const float* W1    = (const float*)d_in[6];
    const float* b1    = (const float*)d_in[7];
    const float* W2    = (const float*)d_in[8];
    const float* b2    = (const float*)d_in[9];
    float* out = (float*)d_out;

    char* ws = (char*)d_ws;
    size_t off = 0;
    auto take = [&](size_t bytes) -> void* {
        void* p = ws + off;
        off += (bytes + 255) & ~(size_t)255;
        return p;
    };
    int*   ix0   = (int*)  take((size_t)NTOK * 4);
    int*   ix1   = (int*)  take((size_t)NTOK * 4);
    float* p0    = (float*)take((size_t)NTOK * 4);
    float* p1    = (float*)take((size_t)NTOK * 4);
    int*   ts0   = (int*)  take((size_t)NTOK * 4);
    int*   ts1   = (int*)  take((size_t)NTOK * 4);
    int*   cnt   = (int*)  take((size_t)NE * 4);
    int*   etok  = (int*)  take((size_t)NE * CAP * 4);
    float* egate = (float*)take((size_t)NE * CAP * 4);
    bf16*  W1b   = (bf16*) take((size_t)NE * HDIM * DDIM * 2);
    bf16*  W2b   = (bf16*) take((size_t)NE * DDIM * HDIM * 2);
    bf16*  Xg    = (bf16*) take((size_t)NE * CAP * DDIM * 2);
    float* oslot = (float*)take((size_t)NE * CAP * DDIM * 4);

    moe_router<<<NTOK / 8, 256, 0, stream>>>(x, noise, Ww, bw, Wn, bn,
                                             ix0, ix1, p0, p1, ts0, ts1);
    moe_cvtw<<<(NE * HDIM * DDIM) / (256 * 8), 256, 0, stream>>>(W1, W2, W1b, W2b);
    moe_scan<<<NE, 1024, 0, stream>>>(ix0, ix1, p0, p1, etok, egate, ts0, ts1, cnt);
    moe_gather_x<<<NE * CAP / 2, 128, 0, stream>>>(x, etok, cnt, Xg);
    moe_ffn<<<NE * (CAP / MT), 512, 0, stream>>>(Xg, W1b, W2b, b1, b2, egate, cnt, oslot);
    moe_gather_out<<<NTOK, 128, 0, stream>>>(oslot, ts0, ts1, out);
}
